// PathModule_15805479649352
// MI455X (gfx1250) — matrix operand order repaired and checked by reading, not compiled, not run on hardware
//
#include <hip/hip_runtime.h>
#include <math.h>
#include <stdint.h>

// ---------------- problem constants (match reference) ----------------
constexpr int kB = 128;     // batch
constexpr int kN = 512;     // nodes
constexpr int kU = 128;     // hidden
constexpr int kD = 16;      // head dim
constexpr int kHD = 128;    // H*D
constexpr int kSteps = 70;  // MAX_STEPS
constexpr int kMObj = 16;
constexpr int kKObj = 4;
constexpr float kInvSqrtD = 0.25f;                 // 1/sqrt(16)
constexpr float kInvSqrtU = 0.08838834764831845f;  // 1/sqrt(128)
constexpr float kClip = 10.0f;

// packed fragment-major weight sizes (halves)
constexpr int kWcatElems = 24 * 4 * 512;   // 128x384  -> 49152
constexpr int kQvpElems  = 8 * 12 * 512;   // 384x128  -> 49152
constexpr int kOvpElems  = 8 * 4 * 512;    // 128x128  -> 16384
constexpr int kQvfWElems = 8 * 4 * 512;    // 128x128  -> 16384

// ---------------- WMMA types / helpers (CDNA5 wave32) ----------------
typedef __attribute__((ext_vector_type(16))) _Float16 v16h;
typedef __attribute__((ext_vector_type(8)))  float    v8f;

// A fragment: 16x32 (MxK) f16 row-major, leading dim lda (LDS or global).
// Lane L<16: M=L, K {0..7,16..23}; L>=16: M=L-16, K {8..15,24..31}.
// Per-lane data = two contiguous 8-half runs -> 2x b128 loads.
__device__ inline v16h load_a_frag(const _Float16* A, int lda, int lane) {
  int m = lane & 15;
  int kb = (lane < 16) ? 0 : 8;
  const _Float16* p0 = A + m * lda + kb;       // K = kb .. kb+7
  const _Float16* p1 = A + m * lda + 16 + kb;  // K = 16+kb .. 16+kb+7
  v16h a;
#pragma unroll
  for (int i = 0; i < 4; ++i) {
    a[2 * i]         = p0[2 * i];
    a[2 * i + 1]     = p0[2 * i + 1];
    a[8 + 2 * i]     = p1[2 * i];
    a[8 + 2 * i + 1] = p1[2 * i + 1];
  }
  return a;
}

// B fragment from fragment-major packed global weights: tile index
// tt*(K/32)+kc selects a 1KB block of 32 lanes x 16 contiguous halves
// (per-lane 32B -> 2x global_load_b128, wave-contiguous, L2-resident).
__device__ inline v16h load_b_frag_g(const _Float16* __restrict__ W, int tile,
                                     int lane) {
  const _Float16* p = W + ((size_t)tile * 32 + lane) * 16;
  v16h b;
#pragma unroll
  for (int e = 0; e < 16; ++e) b[e] = p[e];
  return b;
}

__device__ inline v8f wmma_f16(v16h a, v16h b, v8f c) {
  return __builtin_amdgcn_wmma_f32_16x16x32_f16(
      /*neg_a=*/false, a, /*neg_b=*/false, b,
      /*c_mod=*/(short)0, c, /*reuse_a=*/false, /*reuse_b=*/false);
}

__device__ inline float uhash(unsigned long long x) {
  x += 0x9E3779B97F4A7C15ull;
  x = (x ^ (x >> 30)) * 0xBF58476D1CE4E5B9ull;
  x = (x ^ (x >> 27)) * 0x94D049BB133111EBull;
  x = x ^ (x >> 31);
  unsigned int u24 = (unsigned int)(x >> 40);
  return (u24 + 0.5f) * (1.0f / 16777216.0f);  // (0,1)
}

// =====================================================================
// One-time packing kernels (weights -> fragment-major f16; hvec -> f16)
// =====================================================================
__global__ void convert_hvec_kernel(const float* __restrict__ src,
                                    _Float16* __restrict__ dst, int n) {
  int i = blockIdx.x * blockDim.x + threadIdx.x;
  if (i < n) dst[i] = (_Float16)src[i];
}

// Generic K x C matrix -> fragment-major. element B[k][c]:
//   transpose==0: src[k*C + c]      transpose==1: src[c*K + k]
__global__ void pack_frag_kernel(const float* __restrict__ src,
                                 _Float16* __restrict__ dst, int Kdim,
                                 int Cdim, int transpose, int total) {
  int o = blockIdx.x * blockDim.x + threadIdx.x;
  if (o >= total) return;
  int e = o & 15;
  int L = (o >> 4) & 31;
  int chunk = o >> 9;  // tt*(K/32) + kc
  int kchunks = Kdim >> 5;
  int kc = chunk % kchunks, tt = chunk / kchunks;
  int n = L & 15, kb2 = (L < 16) ? 0 : 8;
  int k = kc * 32 + kb2 + (e & 7) + ((e >> 3) << 4);
  int c = tt * 16 + n;
  float v = transpose ? src[(size_t)c * Kdim + k] : src[(size_t)k * Cdim + c];
  dst[o] = (_Float16)v;
}

// Wcat = [kv_p | vv_p | kvf_W^T] : 128 x 384, fragment-major.
__global__ void pack_wcat_kernel(const float* __restrict__ kv_p,
                                 const float* __restrict__ vv_p,
                                 const float* __restrict__ kvf_W,
                                 _Float16* __restrict__ dst) {
  int o = blockIdx.x * blockDim.x + threadIdx.x;
  if (o >= kWcatElems) return;
  int e = o & 15, L = (o >> 4) & 31, chunk = o >> 9;
  int kc = chunk & 3, tt = chunk >> 2;
  int n = L & 15, kb2 = (L < 16) ? 0 : 8;
  int k = kc * 32 + kb2 + (e & 7) + ((e >> 3) << 4);
  int c = tt * 16 + n;
  float v;
  if (c < 128)      v = kv_p[k * 128 + c];
  else if (c < 256) v = vv_p[k * 128 + (c - 128)];
  else              v = kvf_W[(c - 256) * 128 + k];  // kvf = hvec @ W^T
  dst[o] = (_Float16)v;
}

// =====================================================================
// Kernel 1: loop-invariant projections kv, vv ([B,N,128]) and kvf
// ([B,N,U]). Pure WMMA from global f16 (A: hvec_h rows, B: wcat_h
// fragments). No LDS, no barriers.
// grid = (B*N)/16 row tiles, block = 128 (4 waves x 6 col tiles).
// =====================================================================
__global__ __launch_bounds__(128) void precompute_kernel(
    const _Float16* __restrict__ hvec_h, const _Float16* __restrict__ wcat_h,
    const float* __restrict__ kvf_b, float* __restrict__ kv_ws,
    float* __restrict__ vv_ws, float* __restrict__ kvf_ws) {
  const int row0 = blockIdx.x * 16;  // flat (b*N + n) rows
  const int tid = threadIdx.x, lane = tid & 31, wv = tid >> 5;

  const v8f zero = {0.f, 0.f, 0.f, 0.f, 0.f, 0.f, 0.f, 0.f};
  v8f acc[6] = {zero, zero, zero, zero, zero, zero};
  const _Float16* Abase = hvec_h + (size_t)row0 * kU;

#pragma unroll
  for (int kc = 0; kc < 4; ++kc) {  // K = 128 in chunks of 32
    v16h a = load_a_frag(Abase + kc * 32, kU, lane);
#pragma unroll
    for (int j = 0; j < 6; ++j) {
      int tt = wv * 6 + j;
      v16h b = load_b_frag_g(wcat_h, tt * 4 + kc, lane);
      acc[j] = wmma_f16(a, b, acc[j]);
    }
  }
  // C/D layout: lane n=L&15; vgpr i -> M = i + (L>=16 ? 8 : 0)
  int n = lane & 15;
  int mb = (lane < 16) ? 0 : 8;
#pragma unroll
  for (int j = 0; j < 6; ++j) {
    int c = (wv * 6 + j) * 16 + n;
#pragma unroll
    for (int i = 0; i < 8; ++i) {
      size_t row = (size_t)(row0 + mb + i);
      float v = acc[j][i];
      if (c < 128)      kv_ws[row * 128 + c] = v;
      else if (c < 256) vv_ws[row * 128 + (c - 128)] = v;
      else              kvf_ws[row * 128 + (c - 256)] = v + kvf_b[c - 256];
    }
  }
}

// =====================================================================
// Init: route=0, stats=0, mask state (matches/check) for route[0]=depot.
// =====================================================================
__global__ __launch_bounds__(128) void init_kernel(
    const int* __restrict__ obj_list, const int* __restrict__ hml,
    int* route, int* match_ws, int* check_ws, int* done_ws, float* flp_ws,
    float* len_ws, float* loss_ws) {
  __shared__ int hz_s[128];
  int b = threadIdx.x;
  int hz = 0;
  for (int m = 0; m < kMObj; ++m) hz |= (hml[b * kMObj + m] == 0);
  hz_s[b] = hz;
  __syncthreads();
  for (int s = 64; s > 0; s >>= 1) {
    if (b < s) hz_s[b] |= hz_s[b + s];
    __syncthreads();
  }
  int has_zero = hz_s[0];
  int o0 = obj_list[0], o1 = obj_list[1], o2 = obj_list[2], o3 = obj_list[3];
  int allm = 1;
  for (int m = 0; m < kMObj; ++m) {
    int hv = hml[b * kMObj + m];
    int mm = (hv == o0 || hv == o1 || hv == o2 || hv == o3) ? 1 : 0;
    if (m == 0 && has_zero) mm = 1;
    match_ws[b * kMObj + m] = mm;
    allm &= mm;
  }
  check_ws[b] = allm;
  flp_ws[b] = 0.f;
  len_ws[b] = 0.f;
  for (int i = b; i < (kSteps + 1) * kB; i += 128) route[i] = 0;
  if (b == 0) { *done_ws = 0; *loss_ws = 0.f; }
}

// =====================================================================
// Step kernel A: block = 16 batches, 256 threads (8 waves, 1 col tile
// per wave). vec_concat (prev/first rows streamed via ASYNCcnt-tracked
// global_load_async_to_lds_b128) -> qv (WMMA, B direct from packed
// global) -> split-K masked flash softmax (2 threads per (b,h)) ->
// rec -> hvec_c (WMMA) -> qvf (WMMA + bias) -> global.
// =====================================================================
__global__ __launch_bounds__(256) void step_a_kernel(
    int t, const _Float16* __restrict__ hvec_h, const float* __restrict__ hbar,
    const int* __restrict__ adj, const _Float16* __restrict__ qvp_h,
    const _Float16* __restrict__ ovp_h, const _Float16* __restrict__ qvfW_h,
    const float* __restrict__ qvf_b, const float* __restrict__ vec_1,
    const float* __restrict__ vec_f, const float* __restrict__ kv_ws,
    const float* __restrict__ vv_ws, const int* __restrict__ route,
    const int* __restrict__ check_ws, float* __restrict__ qvf_ws) {
  __shared__ _Float16 sA[16][384];  // A staging (row-major)
  __shared__ float qv_s[16][128];
  __shared__ float rec_s[16][128];
  __shared__ float hc_s[16][128];
  __shared__ float ml_s[2][128];    // split-K softmax merge (m, l)

  const int tid = threadIdx.x, lane = tid & 31, wv = tid >> 5;
  const int b0 = blockIdx.x * 16;
  const v8f zero = {0.f, 0.f, 0.f, 0.f, 0.f, 0.f, 0.f, 0.f};

  // ---- phase 0: vec_concat = [hbar | prev_vec | first_vec] (f16) ----
  for (int i = tid; i < 16 * 128; i += 256) {  // cols 0..127: hbar
    int r = i >> 7, c = i & 127;
    sA[r][c] = (_Float16)hbar[c];
  }
  if (t == 0) {  // cols 128..383 from vec_1 / vec_f (f32 -> f16)
    for (int i = tid; i < 16 * 256; i += 256) {
      int r = i >> 8, c = i & 255;
      int gb = b0 + r;
      float v = (c < 128) ? vec_1[gb * kU + c] : vec_f[gb * kU + (c - 128)];
      sA[r][128 + c] = (_Float16)v;
    }
  } else {
    // cols 128..383: gathered f16 row copies hvec_h[prev]/hvec_h[first]
    // streamed global->LDS via the async engine (no VGPR round-trip).
#pragma unroll
    for (int seg = 0; seg < 2; ++seg) {
      int r = tid >> 4;   // batch row 0..15
      int q = tid & 15;   // 16B chunk within 256B row
      int gb = b0 + r;
      int node = route[(seg == 0 ? (t - 1) : 1) * kB + gb];
      const _Float16* src = hvec_h + ((size_t)gb * kN + node) * kU + q * 8;
      unsigned lds_off =
          (unsigned)(uintptr_t)&sA[r][128 + seg * 128 + q * 8];
      unsigned long long ga = (unsigned long long)(uintptr_t)src;
      asm volatile("global_load_async_to_lds_b128 %0, %1, off"
                   :: "v"(lds_off), "v"(ga)
                   : "memory");
    }
    asm volatile("s_wait_asynccnt 0x0" ::: "memory");
  }
  __syncthreads();

  // ---- phase 1: qv[16,128] = vec_concat[16,384] x qv_p[384,128] ----
  {
    v8f a0 = zero, a1 = zero;  // two chains for WMMA ILP
#pragma unroll
    for (int kc = 0; kc < 12; kc += 2) {
      v16h aa0 = load_a_frag(&sA[0][kc * 32], 384, lane);
      v16h bb0 = load_b_frag_g(qvp_h, wv * 12 + kc, lane);
      a0 = wmma_f16(aa0, bb0, a0);
      v16h aa1 = load_a_frag(&sA[0][(kc + 1) * 32], 384, lane);
      v16h bb1 = load_b_frag_g(qvp_h, wv * 12 + kc + 1, lane);
      a1 = wmma_f16(aa1, bb1, a1);
    }
    v8f r = a0 + a1;
    int n = lane & 15, mb = (lane < 16) ? 0 : 8;
#pragma unroll
    for (int i = 0; i < 8; ++i) qv_s[mb + i][wv * 16 + n] = r[i];
  }
  __syncthreads();

  // ---- phase 2: masked flash softmax, 2 threads per (b,h), split-K ----
  {
    const int bl = tid >> 4, h = (tid >> 1) & 7, p = tid & 1;
    const int gb = b0 + bl;
    const int lastb = route[t * kB + gb];
    const int chk = check_ws[gb];
    const int* adjrow = adj + (size_t)lastb * kN;  // adj_matr[0, last, :]
    float q[kD];
#pragma unroll
    for (int d = 0; d < kD; ++d) q[d] = qv_s[bl][h * kD + d] * kInvSqrtD;
    float m = -3.4e38f, l = 0.f;
    float acc[kD];
#pragma unroll
    for (int d = 0; d < kD; ++d) acc[d] = 0.f;
    const int nlo = p * 256, nhi = nlo + 256;
    for (int n2 = nlo; n2 < nhi; ++n2) {
      const float4* kp =
          (const float4*)(kv_ws + ((size_t)gb * kN + n2) * kHD + h * kD);
      float4 k0 = kp[0], k1 = kp[1], k2 = kp[2], k3 = kp[3];
      float x = q[0] * k0.x + q[1] * k0.y + q[2] * k0.z + q[3] * k0.w +
                q[4] * k1.x + q[5] * k1.y + q[6] * k1.z + q[7] * k1.w +
                q[8] * k2.x + q[9] * k2.y + q[10] * k2.z + q[11] * k2.w +
                q[12] * k3.x + q[13] * k3.y + q[14] * k3.z + q[15] * k3.w;
      if (adjrow[n2] == 0) x = -1e30f;                           // adj mask
      else if ((n2 == 0) ? (chk == 0) : (chk != 0)) x = -1e10f;  // demand mask
      float nm = fmaxf(m, x);
      float sc = __expf(m - nm);
      float pw = __expf(x - nm);
      const float4* vp =
          (const float4*)(vv_ws + ((size_t)gb * kN + n2) * kHD + h * kD);
      float4 v0 = vp[0], v1 = vp[1], v2 = vp[2], v3 = vp[3];
      acc[0] = acc[0] * sc + pw * v0.x;   acc[1] = acc[1] * sc + pw * v0.y;
      acc[2] = acc[2] * sc + pw * v0.z;   acc[3] = acc[3] * sc + pw * v0.w;
      acc[4] = acc[4] * sc + pw * v1.x;   acc[5] = acc[5] * sc + pw * v1.y;
      acc[6] = acc[6] * sc + pw * v1.z;   acc[7] = acc[7] * sc + pw * v1.w;
      acc[8] = acc[8] * sc + pw * v2.x;   acc[9] = acc[9] * sc + pw * v2.y;
      acc[10] = acc[10] * sc + pw * v2.z; acc[11] = acc[11] * sc + pw * v2.w;
      acc[12] = acc[12] * sc + pw * v3.x; acc[13] = acc[13] * sc + pw * v3.y;
      acc[14] = acc[14] * sc + pw * v3.z; acc[15] = acc[15] * sc + pw * v3.w;
      l = l * sc + pw;
      m = nm;
    }
    if (p == 1) {  // publish partial
      ml_s[0][bl * 8 + h] = m;
      ml_s[1][bl * 8 + h] = l;
#pragma unroll
      for (int d = 0; d < kD; ++d) rec_s[bl][h * kD + d] = acc[d];
    }
    __syncthreads();
    if (p == 0) {  // merge halves, normalize
      float m1 = ml_s[0][bl * 8 + h], l1 = ml_s[1][bl * 8 + h];
      float nm = fmaxf(m, m1);
      float e0 = __expf(m - nm), e1 = __expf(m1 - nm);
      float inv = 1.f / (l * e0 + l1 * e1);
#pragma unroll
      for (int d = 0; d < kD; ++d)
        rec_s[bl][h * kD + d] =
            (acc[d] * e0 + rec_s[bl][h * kD + d] * e1) * inv;
    }
  }
  __syncthreads();

  // ---- phase 3: hvec_c[16,128] = rec[16,128] x ov_p[128,128] ----
  for (int i = tid; i < 16 * 128; i += 256) {
    int r = i >> 7, c = i & 127;
    sA[r][c] = (_Float16)rec_s[r][c];
  }
  __syncthreads();
  {
    v8f a0 = zero, a1 = zero;
#pragma unroll
    for (int kc = 0; kc < 4; kc += 2) {
      v16h aa0 = load_a_frag(&sA[0][kc * 32], 384, lane);
      v16h bb0 = load_b_frag_g(ovp_h, wv * 4 + kc, lane);
      a0 = wmma_f16(aa0, bb0, a0);
      v16h aa1 = load_a_frag(&sA[0][(kc + 1) * 32], 384, lane);
      v16h bb1 = load_b_frag_g(ovp_h, wv * 4 + kc + 1, lane);
      a1 = wmma_f16(aa1, bb1, a1);
    }
    v8f r = a0 + a1;
    int n = lane & 15, mb = (lane < 16) ? 0 : 8;
#pragma unroll
    for (int i = 0; i < 8; ++i) hc_s[mb + i][wv * 16 + n] = r[i];
  }
  __syncthreads();

  // ---- phase 4: qvf[16,128] = hvec_c x qvf_W^T + qvf_b -> global ----
  for (int i = tid; i < 16 * 128; i += 256) {
    int r = i >> 7, c = i & 127;
    sA[r][c] = (_Float16)hc_s[r][c];
  }
  __syncthreads();
  {
    v8f a0 = zero, a1 = zero;
#pragma unroll
    for (int kc = 0; kc < 4; kc += 2) {
      v16h aa0 = load_a_frag(&sA[0][kc * 32], 384, lane);
      v16h bb0 = load_b_frag_g(qvfW_h, wv * 4 + kc, lane);
      a0 = wmma_f16(aa0, bb0, a0);
      v16h aa1 = load_a_frag(&sA[0][(kc + 1) * 32], 384, lane);
      v16h bb1 = load_b_frag_g(qvfW_h, wv * 4 + kc + 1, lane);
      a1 = wmma_f16(aa1, bb1, a1);
    }
    v8f r = a0 + a1;
    int n = lane & 15, mb = (lane < 16) ? 0 : 8;
#pragma unroll
    for (int i = 0; i < 8; ++i) {
      int c = wv * 16 + n;
      size_t row = (size_t)(b0 + mb + i);
      qvf_ws[row * 128 + c] = r[i] + qvf_b[c];
    }
  }
}

// =====================================================================
// Step kernel B: one block per batch b. logits = mask(10*tanh(qvf.kvf/
// sqrt(U))); Gumbel-max categorical sample + log_softmax at target.
// =====================================================================
__global__ __launch_bounds__(256) void step_b_kernel(
    int t, const float* __restrict__ kvf_ws, const float* __restrict__ qvf_ws,
    const int* __restrict__ adj, const int* __restrict__ route,
    const int* __restrict__ check_ws, int* __restrict__ target_ws,
    float* __restrict__ logp_ws) {
  __shared__ float qf[128];
  __shared__ float lg[kN];
  __shared__ float redf[256];
  __shared__ int redi[256];
  const int b = blockIdx.x, tid = threadIdx.x;
  if (tid < 128) qf[tid] = qvf_ws[(size_t)b * 128 + tid];
  __syncthreads();
  const int lastb = route[t * kB + b];
  const int chk = check_ws[b];
  const int* adjrow = adj + (size_t)lastb * kN;
  for (int n2 = tid; n2 < kN; n2 += 256) {
    const float4* kp4 = (const float4*)(kvf_ws + ((size_t)b * kN + n2) * kU);
    __builtin_prefetch(kvf_ws + ((size_t)b * kN + n2 + 256) * kU, 0, 1);
    float s = 0.f;
#pragma unroll
    for (int u4 = 0; u4 < 32; ++u4) {
      float4 kk = kp4[u4];
      s += qf[4 * u4 + 0] * kk.x + qf[4 * u4 + 1] * kk.y +
           qf[4 * u4 + 2] * kk.z + qf[4 * u4 + 3] * kk.w;
    }
    s = kClip * tanhf(s * kInvSqrtU);
    if (adjrow[n2] == 0) s = -1e30f;
    else if ((n2 == 0) ? (chk == 0) : (chk != 0)) s = -1e10f;
    lg[n2] = s;
  }
  __syncthreads();
  // max
  redf[tid] = fmaxf(lg[tid], lg[tid + 256]);
  __syncthreads();
  for (int s = 128; s > 0; s >>= 1) {
    if (tid < s) redf[tid] = fmaxf(redf[tid], redf[tid + s]);
    __syncthreads();
  }
  const float maxl = redf[0];
  __syncthreads();
  // sum exp
  redf[tid] = __expf(lg[tid] - maxl) + __expf(lg[tid + 256] - maxl);
  __syncthreads();
  for (int s = 128; s > 0; s >>= 1) {
    if (tid < s) redf[tid] += redf[tid + s];
    __syncthreads();
  }
  const float sumexp = redf[0];
  __syncthreads();
  // Gumbel-max argmax (deterministic hash RNG)
  float bestv = -3.4e38f;
  int besti = 0;
  for (int n2 = tid; n2 < kN; n2 += 256) {
    unsigned long long key =
        ((unsigned long long)(t + 1) << 40) ^ ((unsigned long long)b << 20) ^
        (unsigned long long)n2;
    float u = uhash(key);
    float v = lg[n2] - __logf(-__logf(u));
    if (v > bestv) { bestv = v; besti = n2; }
  }
  redf[tid] = bestv;
  redi[tid] = besti;
  __syncthreads();
  for (int s = 128; s > 0; s >>= 1) {
    if (tid < s && redf[tid + s] > redf[tid]) {
      redf[tid] = redf[tid + s];
      redi[tid] = redi[tid + s];
    }
    __syncthreads();
  }
  if (tid == 0) {
    int tg = redi[0];
    target_ws[b] = tg;
    logp_ws[b] = lg[tg] - maxl - __logf(sumexp);
  }
}

// =====================================================================
// Step kernel C: single block (128 threads, one per batch). Update
// route, matches/check, done flag, flp/length/loss.
// =====================================================================
__global__ __launch_bounds__(128) void step_c_kernel(
    int t, const int* __restrict__ obj_list, const int* __restrict__ hml,
    const float* __restrict__ logp_ws, const int* __restrict__ target_ws,
    int* route, int* match_ws, int* check_ws, int* done_ws, float* flp_ws,
    float* len_ws, float* loss_ws) {
  __shared__ int redi[128];
  __shared__ float redf[128];
  const int b = threadIdx.x;
  const int done = *done_ws;
  int tg = done ? 0 : target_ws[b];
  route[(t + 1) * kB + b] = tg;
  // incremental matches update (route only appends)
  const int o0 = obj_list[tg * kKObj + 0], o1 = obj_list[tg * kKObj + 1];
  const int o2 = obj_list[tg * kKObj + 2], o3 = obj_list[tg * kKObj + 3];
  int allm = 1;
  for (int m = 0; m < kMObj; ++m) {
    int mm = match_ws[b * kMObj + m];
    if (!mm) {
      int hv = hml[b * kMObj + m];
      if (hv == o0 || hv == o1 || hv == o2 || hv == o3) {
        mm = 1;
        match_ws[b * kMObj + m] = 1;
      }
    }
    allm &= mm;
  }
  check_ws[b] = allm;
  // all(target == 0)?
  redi[b] = (tg == 0) ? 1 : 0;
  __syncthreads();
  for (int s = 64; s > 0; s >>= 1) {
    if (b < s) redi[b] &= redi[b + s];
    __syncthreads();
  }
  const int new_done = done | redi[0];
  if (!new_done) {
    flp_ws[b] += logp_ws[b];
    len_ws[b] += (tg != 0) ? 1.f : 0.f;
  }
  __syncthreads();
  redf[b] = flp_ws[b] * len_ws[b];
  __syncthreads();
  for (int s = 64; s > 0; s >>= 1) {
    if (b < s) redf[b] += redf[b + s];
    __syncthreads();
  }
  if (b == 0) {
    if (!new_done) *loss_ws = redf[0] / (float)kB;
    *done_ws = new_done;
  }
}

// =====================================================================
// Finalize: pack outputs (route int32 bit-pattern, loss, flp) into d_out.
// =====================================================================
__global__ void finalize_kernel(const int* __restrict__ route,
                                const float* __restrict__ loss_ws,
                                const float* __restrict__ flp_ws,
                                void* d_out) {
  const int i = blockIdx.x * blockDim.x + threadIdx.x;
  const int RN = (kSteps + 1) * kB;  // 9088
  int* oi = (int*)d_out;
  float* of = (float*)d_out;
  if (i < RN) oi[i] = route[i];
  else if (i == RN) of[i] = *loss_ws;
  else if (i < RN + 1 + kB) of[i] = flp_ws[i - RN - 1];
}

// =====================================================================
extern "C" void kernel_launch(void* const* d_in, const int* in_sizes, int n_in,
                              void* d_out, int out_size, void* d_ws,
                              size_t ws_size, hipStream_t stream) {
  const float* hvec  = (const float*)d_in[0];
  const float* hbar  = (const float*)d_in[1];
  const int*   adj   = (const int*)d_in[2];
  const int*   objl  = (const int*)d_in[3];
  const int*   hml   = (const int*)d_in[4];
  const float* qv_p  = (const float*)d_in[5];
  const float* kv_p  = (const float*)d_in[6];
  const float* vv_p  = (const float*)d_in[7];
  const float* ov_p  = (const float*)d_in[8];
  const float* qvf_W = (const float*)d_in[9];
  const float* qvf_b = (const float*)d_in[10];
  const float* kvf_W = (const float*)d_in[11];
  const float* kvf_b = (const float*)d_in[12];
  const float* vec_1 = (const float*)d_in[13];
  const float* vec_f = (const float*)d_in[14];

  const size_t BNU = (size_t)kB * kN * 128;  // 8.4M elements
  float* kv_ws   = (float*)d_ws;
  float* vv_ws   = kv_ws + BNU;
  float* kvf_ws  = vv_ws + BNU;
  float* qvf_ws  = kvf_ws + BNU;  // B*128
  float* logp_ws = qvf_ws + (size_t)kB * 128;
  float* flp_ws  = logp_ws + kB;
  float* len_ws  = flp_ws + kB;
  float* loss_ws = len_ws + kB;
  int* route     = (int*)(loss_ws + 1);  // (70+1)*B
  int* target_ws = route + (kSteps + 1) * kB;
  int* match_ws  = target_ws + kB;       // B*16
  int* check_ws  = match_ws + kB * kMObj;
  int* done_ws   = check_ws + kB;
  // f16 region, 64B aligned (keeps b128 fragment loads aligned)
  size_t off = (size_t)((char*)(done_ws + 1) - (char*)d_ws);
  off = (off + 63) & ~(size_t)63;
  _Float16* hvec_h = (_Float16*)((char*)d_ws + off);  off += BNU * 2;
  _Float16* wcat_h = (_Float16*)((char*)d_ws + off);  off += kWcatElems * 2;
  _Float16* qvp_h  = (_Float16*)((char*)d_ws + off);  off += kQvpElems * 2;
  _Float16* ovp_h  = (_Float16*)((char*)d_ws + off);  off += kOvpElems * 2;
  _Float16* qvfW_h = (_Float16*)((char*)d_ws + off);  off += kQvfWElems * 2;

  // one-time packing
  convert_hvec_kernel<<<((int)BNU + 255) / 256, 256, 0, stream>>>(
      hvec, hvec_h, (int)BNU);
  pack_wcat_kernel<<<(kWcatElems + 255) / 256, 256, 0, stream>>>(
      kv_p, vv_p, kvf_W, wcat_h);
  pack_frag_kernel<<<(kQvpElems + 255) / 256, 256, 0, stream>>>(
      qv_p, qvp_h, 384, 128, 0, kQvpElems);
  pack_frag_kernel<<<(kOvpElems + 255) / 256, 256, 0, stream>>>(
      ov_p, ovp_h, 128, 128, 0, kOvpElems);
  pack_frag_kernel<<<(kQvfWElems + 255) / 256, 256, 0, stream>>>(
      qvf_W, qvfW_h, 128, 128, 1, kQvfWElems);

  precompute_kernel<<<(kB * kN) / 16, 128, 0, stream>>>(
      hvec_h, wcat_h, kvf_b, kv_ws, vv_ws, kvf_ws);
  init_kernel<<<1, 128, 0, stream>>>(objl, hml, route, match_ws, check_ws,
                                     done_ws, flp_ws, len_ws, loss_ws);
  for (int t = 0; t < kSteps; ++t) {
    step_a_kernel<<<kB / 16, 256, 0, stream>>>(
        t, hvec_h, hbar, adj, qvp_h, ovp_h, qvfW_h, qvf_b, vec_1, vec_f,
        kv_ws, vv_ws, route, check_ws, qvf_ws);
    step_b_kernel<<<kB, 256, 0, stream>>>(t, kvf_ws, qvf_ws, adj, route,
                                          check_ws, target_ws, logp_ws);
    step_c_kernel<<<1, 128, 0, stream>>>(t, objl, hml, logp_ws, target_ws,
                                         route, match_ws, check_ws, done_ws,
                                         flp_ws, len_ws, loss_ws);
  }
  const int total = (kSteps + 1) * kB + 1 + kB;
  finalize_kernel<<<(total + 255) / 256, 256, 0, stream>>>(route, loss_ws,
                                                           flp_ws, d_out);
}
